// RPNTargetSampler_53145925320989
// MI455X (gfx1250) — compile-verified, matrix operand order checked by reading, not run on hardware
//
#include <hip/hip_runtime.h>
#include <hip/hip_bf16.h>

#define B   8
#define NA  200000
#define K   20
#define HB  65536

typedef __attribute__((ext_vector_type(16))) _Float16 v16h;
typedef __attribute__((ext_vector_type(8)))  float    v8f;

// ---- workspace layout (u32 words) ----
#define GTMAX_OFF 0
#define FGH_OFF   (GTMAX_OFF + B*K)
#define BGH_OFF   (FGH_OFF + B*HB)
#define FGT_OFF   (BGH_OFF + B*HB)
#define BGT_OFF   (FGT_OFF + B)
#define WS_WORDS  (BGT_OFF + B)

__device__ __forceinline__ unsigned pcg_hash(unsigned x) {
  x = x * 747796405u + 2891336453u;
  unsigned w = ((x >> ((x >> 28u) + 4u)) ^ x) * 277803737u;
  return (w >> 22u) ^ w;
}
__device__ __forceinline__ unsigned rbucket(int b, int n) {
  return pcg_hash((unsigned)n * 2654435761u + (unsigned)b * 40503u) >> 16;
}

// IoU with the reference's degenerate-box rules baked in.
__device__ __forceinline__ float iou_ref(float ax1, float ay1, float ax2, float ay2,
                                         float gx1, float gy1, float gx2, float gy2,
                                         float garea, int gzero) {
  float aw = ax2 - ax1 + 1.0f, ah = ay2 - ay1 + 1.0f;
  float aarea = aw * ah;
  float iw = fminf(ax2, gx2) - fmaxf(ax1, gx1) + 1.0f;
  float ih = fminf(ay2, gy2) - fmaxf(ay1, gy1) + 1.0f;
  float inter = fmaxf(iw, 0.0f) * fmaxf(ih, 0.0f);
  float ov = inter / (aarea + garea - inter);
  if (gzero) ov = 0.0f;
  if (aw == 1.0f && ah == 1.0f) ov = -1.0f;
  return ov;
}

__global__ void k_zero(unsigned* __restrict__ ws) {
  int i = blockIdx.x * blockDim.x + threadIdx.x;
  if (i < WS_WORDS) ws[i] = 0u;
}

// Pass A: gt_max[b,k] = max over anchors of IoU (clamped at 0; bit-monotone u32 atomics)
__global__ void k_gtmax(const float* __restrict__ anchors,
                        const float* __restrict__ gt,
                        unsigned* __restrict__ gmax) {
  __shared__ float sg[K][4];
  __shared__ float sga[K];
  __shared__ int   sgz[K];
  __shared__ unsigned smax[K];
  int b = blockIdx.y, tid = threadIdx.x;
  if (tid < K) {
    const float* g = gt + (size_t)(b * K + tid) * 5;
    float x1 = g[0], y1 = g[1], x2 = g[2], y2 = g[3];
    sg[tid][0] = x1; sg[tid][1] = y1; sg[tid][2] = x2; sg[tid][3] = y2;
    float gw = x2 - x1 + 1.0f, gh = y2 - y1 + 1.0f;
    sga[tid] = gw * gh;
    sgz[tid] = (gw == 1.0f && gh == 1.0f) ? 1 : 0;
    smax[tid] = 0u;
  }
  __syncthreads();
  int n = blockIdx.x * blockDim.x + tid;
  if (n < NA) {
    const float4 a = ((const float4*)anchors)[n];
    #pragma unroll
    for (int k = 0; k < K; ++k) {
      float ov = iou_ref(a.x, a.y, a.z, a.w,
                         sg[k][0], sg[k][1], sg[k][2], sg[k][3], sga[k], sgz[k]);
      atomicMax(&smax[k], __float_as_uint(fmaxf(ov, 0.0f)));
    }
  }
  __syncthreads();
  if (tid < K) atomicMax(&gmax[b * K + tid], smax[tid]);
}

// Pass B: labels (pre-subsample) + matches + fg/bg histograms.
// Wave owns 16 anchors x 32 K-slots in the exact 16-bit A-matrix layout:
//   lanes 0-15 : row M=lane,    K = 0..7 then 16..23  (slots j=0..15)
//   lanes16-31 : row M=lane-16, K = 8..15 then 24..31
// keep-count = A(16x32 indicator, f16) x ones(32x16) via v_wmma_f32_16x16x32_f16.
__global__ void k_labels(const float* __restrict__ anchors,
                         const float* __restrict__ gt,
                         const unsigned* __restrict__ gmax,
                         unsigned* __restrict__ fgh,
                         unsigned* __restrict__ bgh,
                         float* __restrict__ labels,
                         float* __restrict__ matches) {
  __shared__ float sg[K][4];
  __shared__ float sga[K];
  __shared__ int   sgz[K];
  __shared__ float sadj[K];
  int b = blockIdx.y, tid = threadIdx.x;
  if (tid < K) {
    const float* g = gt + (size_t)(b * K + tid) * 5;
    float x1 = g[0], y1 = g[1], x2 = g[2], y2 = g[3];
    sg[tid][0] = x1; sg[tid][1] = y1; sg[tid][2] = x2; sg[tid][3] = y2;
    float gw = x2 - x1 + 1.0f, gh = y2 - y1 + 1.0f;
    sga[tid] = gw * gh;
    sgz[tid] = (gw == 1.0f && gh == 1.0f) ? 1 : 0;
    float gm = __uint_as_float(gmax[b * K + tid]);
    sadj[tid] = (gm == 0.0f) ? 1e-5f : gm;
  }
  __syncthreads();

  int lane = tid & 31, wave = tid >> 5;
  int m = lane & 15;
  int n = blockIdx.x * 128 + wave * 16 + m;
  bool valid = (n < NA);
  float4 a = valid ? ((const float4*)anchors)[n] : make_float4(0.f, 0.f, 0.f, 0.f);
  {  // stream-ahead prefetch of anchors (global_prefetch_b8)
    int pn = n + 4096; if (pn >= NA) pn = NA - 1;
    __builtin_prefetch(anchors + (size_t)pn * 4, 0, 0);
  }

  int kofs = (lane & 16) ? 8 : 0;
  float pmax = -2.0f; int parg = 0;
  v16h A;
  #pragma unroll
  for (int j = 0; j < 16; ++j) {
    int k = ((j < 8) ? j : j + 8) + kofs;
    _Float16 ind = (_Float16)0.0f;
    if (k < K && valid) {
      float ov = iou_ref(a.x, a.y, a.z, a.w,
                         sg[k][0], sg[k][1], sg[k][2], sg[k][3], sga[k], sgz[k]);
      if (ov > pmax) { pmax = ov; parg = k; }
      ind = (ov == sadj[k]) ? (_Float16)1.0f : (_Float16)0.0f;
    }
    A[j] = ind;
  }
  // combine the two half-lanes that share anchor m: full max/argmax over K
  float omax = __shfl_xor(pmax, 16, 32);
  int   oarg = __shfl_xor(parg, 16, 32);
  if (omax > pmax || (omax == pmax && oarg < parg)) { pmax = omax; parg = oarg; }

  // keep-count via WMMA: D[m, n] = sum_k A[m,k]  (B = all-ones)
  v16h Bm;
  #pragma unroll
  for (int j = 0; j < 16; ++j) Bm[j] = (_Float16)1.0f;
  v8f C = {};
  v8f D = __builtin_amdgcn_wmma_f32_16x16x32_f16(false, A, false, Bm,
                                                 (short)0, C, false, false);
  // lanes 0-15 hold rows M=0..7 in D[0..7]; lanes 16-31 hold rows M=8..15
  int idx = m & 7;
  float contrib = 0.0f;
  bool have = (lane < 16) ? (m < 8) : (m >= 8);
  if (have) {
    float r = D[0];
    #pragma unroll
    for (int i = 1; i < 8; ++i) r = (idx == i) ? D[i] : r;
    contrib = r;
  }
  float keepv = contrib + __shfl_xor(contrib, 16, 32);

  if (valid) {
    float lab = -1.0f;
    if (pmax < 0.3f) lab = 0.0f;                    // RPN_NEGATIVE_OVERLAP
    if (keepv > 0.5f || pmax >= 0.7f) lab = 1.0f;   // gt-max tie or RPN_POSITIVE_OVERLAP
    labels[(size_t)b * NA + n]  = lab;
    matches[(size_t)b * NA + n] = (float)(parg + b * K);
    unsigned h = rbucket(b, n);
    if (lab == 1.0f)      atomicAdd(&fgh[(size_t)b * HB + h], 1u);
    else if (lab == 0.0f) atomicAdd(&bgh[(size_t)b * HB + h], 1u);
  }
}

// Pass C: per-batch rank-select threshold buckets (fg target=128, bg target=256-fg_kept)
__global__ void k_thresh(const unsigned* __restrict__ fgh,
                         const unsigned* __restrict__ bgh,
                         unsigned* __restrict__ fgT,
                         unsigned* __restrict__ bgT) {
  __shared__ unsigned csum[256];
  __shared__ unsigned sFgKept;
  int b = blockIdx.x, t = threadIdx.x;
  for (int pass = 0; pass < 2; ++pass) {
    const unsigned* H = (pass == 0 ? fgh : bgh) + (size_t)b * HB;
    unsigned c = 0;
    int base = t * 256;
    for (int i = 0; i < 256; ++i) c += H[base + i];
    csum[t] = c;
    __syncthreads();
    if (t == 0) {
      unsigned total = 0;
      for (int i = 0; i < 256; ++i) total += csum[i];
      unsigned target = (pass == 0) ? 128u : (256u - sFgKept);
      unsigned T = HB - 1;
      if (total > target) {
        unsigned cum = 0; int Cc = 0;
        for (int i = 0; i < 256; ++i) {
          if (cum + csum[i] >= target) { Cc = i; break; }
          cum += csum[i];
        }
        unsigned Tt = (unsigned)Cc * 256u;
        for (int j = 0; j < 256; ++j) {
          cum += H[Cc * 256 + j];
          if (cum >= target) { Tt = (unsigned)(Cc * 256 + j); break; }
        }
        T = Tt;
      }
      if (pass == 0) { sFgKept = (total < 128u) ? total : 128u; fgT[b] = T; }
      else           { bgT[b] = T; }
    }
    __syncthreads();
  }
}

// Pass D: apply deterministic subsampling in place.
__global__ void k_subsample(float* __restrict__ labels,
                            const unsigned* __restrict__ fgT,
                            const unsigned* __restrict__ bgT) {
  int b = blockIdx.y;
  int n = blockIdx.x * blockDim.x + threadIdx.x;
  if (n >= NA) return;
  float lab = labels[(size_t)b * NA + n];
  unsigned h = rbucket(b, n);
  if (lab == 1.0f)      { if (h > fgT[b]) lab = -1.0f; }
  else if (lab == 0.0f) { if (h > bgT[b]) lab = -1.0f; }
  labels[(size_t)b * NA + n] = lab;
}

// Pass E: inds_inside = arange(N); anchors pass-through copy.
__global__ void k_copy(const float* __restrict__ anchors,
                       float* __restrict__ inds,
                       float* __restrict__ aout) {
  int n = blockIdx.x * blockDim.x + threadIdx.x;
  if (n >= NA) return;
  inds[n] = (float)n;
  ((float4*)aout)[n] = ((const float4*)anchors)[n];
}

extern "C" void kernel_launch(void* const* d_in, const int* in_sizes, int n_in,
                              void* d_out, int out_size, void* d_ws, size_t ws_size,
                              hipStream_t stream) {
  (void)in_sizes; (void)n_in; (void)out_size; (void)ws_size;
  // inputs: [0] rpn_cls_score (unused) [1] gt_boxes (B,K,5) [2] im_info (unused)
  //         [3] num_boxes (unused)     [4] anchors (1,N,4)
  const float* gt      = (const float*)d_in[1];
  const float* anchors = (const float*)d_in[4];

  float* out     = (float*)d_out;
  float* labels  = out;                                   // B*NA
  float* matches = out + (size_t)B * NA;                  // B*NA
  float* inds    = out + (size_t)2 * B * NA;              // NA
  float* aout    = out + (size_t)2 * B * NA + NA;         // NA*4

  unsigned* ws   = (unsigned*)d_ws;
  unsigned* gmax = ws + GTMAX_OFF;
  unsigned* fgh  = ws + FGH_OFF;
  unsigned* bgh  = ws + BGH_OFF;
  unsigned* fgT  = ws + FGT_OFF;
  unsigned* bgT  = ws + BGT_OFF;

  k_zero<<<(WS_WORDS + 255) / 256, 256, 0, stream>>>(ws);
  k_gtmax<<<dim3((NA + 255) / 256, B), 256, 0, stream>>>(anchors, gt, gmax);
  k_labels<<<dim3((NA + 127) / 128, B), 256, 0, stream>>>(anchors, gt, gmax,
                                                          fgh, bgh, labels, matches);
  k_thresh<<<B, 256, 0, stream>>>(fgh, bgh, fgT, bgT);
  k_subsample<<<dim3((NA + 255) / 256, B), 256, 0, stream>>>(labels, fgT, bgT);
  k_copy<<<(NA + 255) / 256, 256, 0, stream>>>(anchors, inds, aout);
}
